// SelfAttention_23622320128116
// MI455X (gfx1250) — matrix operand order repaired and checked by reading, not compiled, not run on hardware
//
#include <hip/hip_runtime.h>

// ---------------------------------------------------------------------------
// Self-attention + MaxPool12 + linear head for MI455X (gfx1250, wave32).
// All heavy math in v_wmma_f32_16x16x32_bf16 (f32 accumulate).
// B=32, N=1536, D=1024.  Flash-style softmax, pool fused (48 = lcm(12,16)).
// QKV GEMM stages weight tiles into LDS with GLOBAL_LOAD_ASYNC_TO_LDS_B128
// (double buffered on ASYNCcnt).
// ---------------------------------------------------------------------------

typedef __attribute__((ext_vector_type(16))) __bf16 v16bf;
typedef __attribute__((ext_vector_type(8)))  float  v8f;
typedef __attribute__((ext_vector_type(4)))  int    v4i;
typedef __attribute__((address_space(1))) v4i g_v4i;   // global int4
typedef __attribute__((address_space(3))) v4i l_v4i;   // LDS int4

#define B_    32
#define N_    1536
#define D_    1024
#define POOL_ 12
#define NP_   128   // N_/POOL_

#define WMMA_BF16(A, Bm, C) \
  __builtin_amdgcn_wmma_f32_16x16x32_bf16(false, (A), false, (Bm), (short)0, (C), false, false)

#if __has_builtin(__builtin_amdgcn_global_load_async_to_lds_b128)
#define HAVE_ASYNC_LDS 1
#else
#define HAVE_ASYNC_LDS 0
#endif

template <int Ncnt>
__device__ __forceinline__ void wait_asynccnt() {
#if __has_builtin(__builtin_amdgcn_s_wait_asynccnt)
  __builtin_amdgcn_s_wait_asynccnt(Ncnt);
#else
  asm volatile("s_wait_asynccnt %0" ::"i"(Ncnt) : "memory");
#endif
}

__device__ __forceinline__ __bf16 f2bf(float f) {
  unsigned u = __builtin_bit_cast(unsigned, f);
  unsigned r = u + 0x7FFFu + ((u >> 16) & 1u);   // round-to-nearest-even
  unsigned short h = (unsigned short)(r >> 16);
  return __builtin_bit_cast(__bf16, h);
}

// A fragment: 16(M) x 32(K) bf16 tile, row-major source with leading dim ld.
// Contiguous K order (same as load_frag_b): lane L -> M = L%16; VGPR p holds K pair at (L/16)*16 + p*2.
__device__ __forceinline__ v16bf load_frag_a(const __bf16* base, int ld, int lane) {
  const int m = lane & 15, half = lane >> 4;
  const __bf16* rp = base + (size_t)m * ld;
  v16bf a;
#pragma unroll
  for (int p = 0; p < 8; ++p) {
    int k = half * 16 + p * 2;
    a[2 * p]     = rp[k];
    a[2 * p + 1] = rp[k + 1];
  }
  return a;
}

// B fragment: 32(K) x 16(N); column n of B is contiguous over K at base + n*ld.
// Same contiguous K order as A: lane L -> N = L%16, 16 contiguous K starting at (L/16)*16.
__device__ __forceinline__ v16bf load_frag_b(const __bf16* base, int ld, int lane) {
  const int n = lane & 15, half = lane >> 4;
  const __bf16* cp = base + (size_t)n * ld + half * 16;
  v16bf b;
#pragma unroll
  for (int p = 0; p < 16; ++p) b[p] = cp[p];
  return b;
}

// ---------------------------------------------------------------------------
// Kernel 1: f32 -> bf16 conversion (vectorized x4)
// ---------------------------------------------------------------------------
struct alignas(8) bf16x4 { __bf16 v[4]; };

__global__ void cvt_f32_to_bf16_kernel(const float4* __restrict__ src,
                                       bf16x4* __restrict__ dst, int n4) {
  int i = blockIdx.x * blockDim.x + threadIdx.x;
  if (i < n4) {
    float4 f = src[i];
    bf16x4 o;
    o.v[0] = f2bf(f.x); o.v[1] = f2bf(f.y);
    o.v[2] = f2bf(f.z); o.v[3] = f2bf(f.w);
    dst[i] = o;
  }
}

// ---------------------------------------------------------------------------
// Kernel 2: fused QKV projection.  grid = (384, 16, 3), block = 256 (8 waves).
// Block computes a 128x64 tile of q/k/v = x @ W^T.  The 64x32 bf16 weight
// tile is staged into LDS (async global->LDS, double buffered); A fragments
// come straight from global (b128 clauses).  z==2 (v) stores transposed per
// batch: vtb[b][d][n]  (so P@V B-fragments are contiguous over m later).
// ---------------------------------------------------------------------------
__global__ void __launch_bounds__(256)
qkv_gemm_kernel(const __bf16* __restrict__ xb,
                const __bf16* __restrict__ wqb,
                const __bf16* __restrict__ wkb,
                const __bf16* __restrict__ wvb,
                __bf16* __restrict__ qb,
                __bf16* __restrict__ kb,
                __bf16* __restrict__ vtb) {
  __shared__ __bf16 wtile[2][64 * 32];   // 2 x 4 KB, [out-col][k] row-major

  const int tid  = threadIdx.x;
  const int lane = tid & 31;
  const int wave = tid >> 5;
  const int wr = wave & 3, wc = wave >> 2;
  const int z = blockIdx.z;
  const int colBlk  = blockIdx.y * 64;              // block's first out column
  const int rowBase = blockIdx.x * 128 + wr * 32;   // global row in [0, B*N)
  const __bf16* W = (z == 0) ? wqb : (z == 1) ? wkb : wvb;

  // each thread stages 16 bytes of the 64x32 weight tile per k-step
  const int srow = tid >> 2;        // 0..63  (output column within block)
  const int sseg = tid & 3;         // 4 x 8 bf16 segments per row
  const __bf16* sg0 = W + (size_t)(colBlk + srow) * D_ + sseg * 8;

  v8f acc[2][2] = {};
  const __bf16* a0p = xb + (size_t)rowBase * D_;
  const __bf16* a1p = a0p + (size_t)16 * D_;

#if HAVE_ASYNC_LDS
#define STAGE_W(KK, BUF)                                                     \
  __builtin_amdgcn_global_load_async_to_lds_b128(                            \
      (g_v4i*)(sg0 + (KK)),                                                  \
      (l_v4i*)&wtile[(BUF)][srow * 32 + sseg * 8], 0, 0)
#else
#define STAGE_W(KK, BUF)                                                     \
  *(uint4*)&wtile[(BUF)][srow * 32 + sseg * 8] = *(const uint4*)(sg0 + (KK))
#endif

  STAGE_W(0, 0);
  for (int kk = 0, it = 0; kk < D_; kk += 32, ++it) {
    const int buf = it & 1;
    const bool more = (kk + 32) < D_;
    if (more) STAGE_W(kk + 32, buf ^ 1);
#if HAVE_ASYNC_LDS
    if (more) wait_asynccnt<1>();    // current tile's transfers complete
    else      wait_asynccnt<0>();
#endif
    __syncthreads();

    __builtin_prefetch(a0p + kk + 128, 0, 1);
    v16bf a0 = load_frag_a(a0p + kk, D_, lane);
    v16bf a1 = load_frag_a(a1p + kk, D_, lane);
    const __bf16* bt = &wtile[buf][(wc * 32) * 32];
    v16bf b0 = load_frag_b(bt, 32, lane);
    v16bf b1 = load_frag_b(bt + 16 * 32, 32, lane);
    acc[0][0] = WMMA_BF16(a0, b0, acc[0][0]);
    acc[0][1] = WMMA_BF16(a0, b1, acc[0][1]);
    acc[1][0] = WMMA_BF16(a1, b0, acc[1][0]);
    acc[1][1] = WMMA_BF16(a1, b1, acc[1][1]);
    __syncthreads();                 // buffer consumed; next stage may overwrite
  }
#undef STAGE_W

  const int colBase = colBlk + wc * 32;
  const int bi = (blockIdx.x * 128) / N_;   // whole 128-row block is in one batch
  const int n  = lane & 15, half = lane >> 4;
  if (z == 2) {
    const int n0 = rowBase - bi * N_;
#pragma unroll
    for (int ti = 0; ti < 2; ++ti)
#pragma unroll
      for (int tj = 0; tj < 2; ++tj)
#pragma unroll
        for (int i = 0; i < 8; ++i)
          vtb[((size_t)bi * D_ + colBase + tj * 16 + n) * N_ +
              (n0 + ti * 16 + half * 8 + i)] = f2bf(acc[ti][tj][i]);
  } else {
    __bf16* dst = (z == 0) ? qb : kb;
#pragma unroll
    for (int ti = 0; ti < 2; ++ti)
#pragma unroll
      for (int tj = 0; tj < 2; ++tj)
#pragma unroll
        for (int i = 0; i < 8; ++i)
          dst[(size_t)(rowBase + ti * 16 + half * 8 + i) * D_ +
              (colBase + tj * 16 + n)] = f2bf(acc[ti][tj][i]);
  }
}

// ---------------------------------------------------------------------------
// Kernel 3: flash attention + fused MaxPool12.
// grid = (N/48 = 32, B = 32, 2);  block = 256 (8 waves); dyn LDS = 43776 B.
// Each block: 48 query rows x 512 V columns.  Wave w owns V cols [w*64,w*64+64).
// ---------------------------------------------------------------------------
__global__ void __launch_bounds__(256)
attn_pool_kernel(const __bf16* __restrict__ qb,
                 const __bf16* __restrict__ kb,
                 const __bf16* __restrict__ vtb,
                 float* __restrict__ pooled) {
  extern __shared__ char smem[];
  float*  S    = (float*)smem;                               // 48*64 f32  (12288 B)
  __bf16* Pb   = (__bf16*)(smem + 48 * 64 * 4);              // 48*64 bf16 ( 6144 B)
  float*  mrow = (float*)(smem + 48 * 64 * 4 + 48 * 64 * 2); // 48
  float*  lrow = mrow + 48;
  float*  crow = lrow + 48;
  float*  attS = (float*)(smem + 19200);                     // 2 * 48*64 f32 (24576 B)

  const int tid  = threadIdx.x;
  const int lane = tid & 31;
  const int wave = tid >> 5;
  const int nb = blockIdx.x;            // 48-row block index
  const int b  = blockIdx.y;
  const int vs = blockIdx.z;            // which half of D_V
  const int q0 = nb * 48;
  const int vbase = vs * 512 + wave * 64;
  const float norm = 0.03125f;          // 1/sqrt(1024)

  if (tid < 48) { mrow[tid] = -3.0e38f; lrow[tid] = 0.0f; }
  v8f acc[3][4] = {};
  __syncthreads();

  for (int m0 = 0; m0 < N_; m0 += 64) {
    // ---- Phase A: waves 0..2 compute S(16x64) = Q Kt for their row tile ----
    if (wave < 3) {
      const __bf16* qp = qb + (size_t)(b * N_ + q0 + wave * 16) * D_;
#pragma unroll
      for (int mt = 0; mt < 4; ++mt) {
        const __bf16* kp = kb + (size_t)(b * N_ + m0 + mt * 16) * D_;
        __builtin_prefetch(kp + 512, 0, 1);
        v8f s = {};
        for (int kk = 0; kk < D_; kk += 32) {
          v16bf af = load_frag_a(qp + kk, D_, lane);
          v16bf bf = load_frag_b(kp + kk, D_, lane);
          s = WMMA_BF16(af, bf, s);
        }
        int n = lane & 15, half = lane >> 4;
#pragma unroll
        for (int i = 0; i < 8; ++i)
          S[(wave * 16 + half * 8 + i) * 64 + mt * 16 + n] = s[i] * norm;
      }
    }
    __syncthreads();

    // ---- Phase B: online softmax, one thread per row ----
    if (tid < 48) {
      float mOld = mrow[tid], mNew = mOld;
      const float* sr = S + tid * 64;
#pragma unroll 8
      for (int j = 0; j < 64; ++j) mNew = fmaxf(mNew, sr[j]);
      float c = __expf(mOld - mNew);
      float l = lrow[tid] * c;
      __bf16* pr = Pb + tid * 64;
#pragma unroll 8
      for (int j = 0; j < 64; ++j) {
        float p = __expf(sr[j] - mNew);
        l += p;
        pr[j] = f2bf(p);
      }
      mrow[tid] = mNew; lrow[tid] = l; crow[tid] = c;
    }
    __syncthreads();

    // ---- Phase C: acc = acc * c + P @ V (all 8 waves) ----
    {
      const int half = lane >> 4;
#pragma unroll
      for (int rt = 0; rt < 3; ++rt) {
        float cf[8];
#pragma unroll
        for (int i = 0; i < 8; ++i) cf[i] = crow[rt * 16 + half * 8 + i];
#pragma unroll
        for (int ct = 0; ct < 4; ++ct) {
#pragma unroll
          for (int i = 0; i < 8; ++i) acc[rt][ct][i] *= cf[i];
#pragma unroll
          for (int mc = 0; mc < 2; ++mc) {
            v16bf af = load_frag_a(Pb + rt * 16 * 64 + mc * 32, 64, lane);
            v16bf bf = load_frag_b(
                vtb + ((size_t)b * D_ + vbase + ct * 16) * N_ + (m0 + mc * 32),
                N_, lane);
            acc[rt][ct] = WMMA_BF16(af, bf, acc[rt][ct]);
          }
        }
      }
    }
    __syncthreads();
  }

  // ---- Finalize: normalize, stage through LDS 2 waves at a time, max-pool ----
  const int n = lane & 15, half = lane >> 4;
  for (int pp = 0; pp < 4; ++pp) {
    if ((wave >> 1) == pp) {
      const int wreg = wave & 1;
#pragma unroll
      for (int rt = 0; rt < 3; ++rt) {
        float inv[8];
#pragma unroll
        for (int i = 0; i < 8; ++i) inv[i] = 1.0f / lrow[rt * 16 + half * 8 + i];
#pragma unroll
        for (int ct = 0; ct < 4; ++ct)
#pragma unroll
          for (int i = 0; i < 8; ++i)
            attS[wreg * (48 * 64) + (rt * 16 + half * 8 + i) * 64 + ct * 16 + n] =
                acc[rt][ct][i] * inv[i];
      }
    }
    __syncthreads();
    for (int o = tid; o < 4 * 128; o += 256) {
      int p = o >> 7;            // pool window within the 48 rows (0..3)
      int cc = o & 127;          // column within this pass's 128
      int w = cc >> 6, c2 = cc & 63;
      const float* ap = attS + w * (48 * 64) + c2;
      float mx = -3.0e38f;
#pragma unroll
      for (int r = 0; r < POOL_; ++r) mx = fmaxf(mx, ap[(p * POOL_ + r) * 64]);
      pooled[((size_t)b * NP_ + nb * 4 + p) * D_ + vs * 512 + pp * 128 + cc] = mx;
    }
    __syncthreads();
  }
}

// ---------------------------------------------------------------------------
// Kernel 4: head projection  out[b][c][p] = pooled[b][p][:] . W_head[c][:] + b[c]
// ---------------------------------------------------------------------------
__global__ void head_kernel(const float* __restrict__ pooled,
                            const float* __restrict__ Wh,
                            const float* __restrict__ bh,
                            float* __restrict__ out) {
  int idx = blockIdx.x * blockDim.x + threadIdx.x;  // B*2*NP = 8192
  int b = idx >> 8, c = (idx >> 7) & 1, p = idx & 127;
  const float4* pr = (const float4*)(pooled + ((size_t)b * NP_ + p) * D_);
  const float4* wr = (const float4*)(Wh + (size_t)c * D_);
  float s = 0.0f;
  for (int d = 0; d < D_ / 4; ++d) {
    float4 a = pr[d], w = wr[d];
    s += a.x * w.x + a.y * w.y + a.z * w.z + a.w * w.w;
  }
  out[idx] = s + bh[c];
}

// ---------------------------------------------------------------------------
// Launcher
// ---------------------------------------------------------------------------
extern "C" void kernel_launch(void* const* d_in, const int* in_sizes, int n_in,
                              void* d_out, int out_size, void* d_ws, size_t ws_size,
                              hipStream_t stream) {
  const float* x  = (const float*)d_in[0];
  const float* Wq = (const float*)d_in[1];
  const float* Wk = (const float*)d_in[2];
  const float* Wv = (const float*)d_in[3];
  const float* Wh = (const float*)d_in[4];
  const float* bh = (const float*)d_in[5];
  float* out = (float*)d_out;

  const size_t XE = (size_t)B_ * N_ * D_;   // 50,331,648 elems
  const size_t WE = (size_t)D_ * D_;        //  1,048,576 elems

  __bf16* xb  = (__bf16*)d_ws;
  __bf16* qb  = xb  + XE;
  __bf16* kb  = qb  + XE;
  __bf16* vtb = kb  + XE;       // v transposed per batch: [B][D][N]
  __bf16* wqb = vtb + XE;
  __bf16* wkb = wqb + WE;
  __bf16* wvb = wkb + WE;
  float*  pooled = (float*)(wvb + WE);      // [B][NP][D] f32

  // f32 -> bf16
  cvt_f32_to_bf16_kernel<<<(int)(XE / 4 / 256), 256, 0, stream>>>(
      (const float4*)x, (bf16x4*)xb, (int)(XE / 4));
  cvt_f32_to_bf16_kernel<<<(int)(WE / 4 / 256), 256, 0, stream>>>(
      (const float4*)Wq, (bf16x4*)wqb, (int)(WE / 4));
  cvt_f32_to_bf16_kernel<<<(int)(WE / 4 / 256), 256, 0, stream>>>(
      (const float4*)Wk, (bf16x4*)wkb, (int)(WE / 4));
  cvt_f32_to_bf16_kernel<<<(int)(WE / 4 / 256), 256, 0, stream>>>(
      (const float4*)Wv, (bf16x4*)wvb, (int)(WE / 4));

  // q, k, v projections (v stored transposed)
  qkv_gemm_kernel<<<dim3((B_ * N_) / 128, D_ / 64, 3), 256, 0, stream>>>(
      xb, wqb, wkb, wvb, qb, kb, vtb);

  // flash attention + fused MaxPool12
  attn_pool_kernel<<<dim3(N_ / 48, B_, 2), 256, 43776, stream>>>(qb, kb, vtb, pooled);

  // linear head
  head_kernel<<<(B_ * 2 * NP_) / 256, 256, 0, stream>>>(pooled, Wh, bh, out);
}
